// WKVLayer_31860067402034
// MI455X (gfx1250) — matrix operand order repaired and checked by reading, not compiled, not run on hardware
//
#include <hip/hip_runtime.h>
#include <hip/hip_bf16.h>

// ---------------------------------------------------------------------------
// Types for WMMA fragments (gfx1250 / CDNA5, wave32)
// ---------------------------------------------------------------------------
typedef __bf16 v8bf  __attribute__((ext_vector_type(8)));
typedef __bf16 v16bf __attribute__((ext_vector_type(16)));
typedef float  v8f   __attribute__((ext_vector_type(8)));

#define S_DIM 2048
#define B_DIM 8
#define E_DIM 1024
#define M_DIM (S_DIM * B_DIM)     // 16384 rows for all three GEMMs
#define NCHAN (B_DIM * E_DIM)     // 8192 scan channels
#define NCHUNK 32                 // chunks along S for the parallel scan
#define CLEN  (S_DIM / NCHUNK)    // 64 steps per chunk

// ---------------------------------------------------------------------------
// transpose + convert fp32 W (K x N) -> bf16 W^T (N x K)
// ---------------------------------------------------------------------------
__global__ void wt_convert_kernel(const float* __restrict__ W,
                                  __bf16* __restrict__ Wt, int E) {
    int i = blockIdx.x * blockDim.x + threadIdx.x;
    if (i >= E * E) return;
    int n = i / E;
    int k = i - n * E;
    Wt[(size_t)n * E + k] = (__bf16)W[(size_t)k * E + n];
}

// ---------------------------------------------------------------------------
// token-mix prep: xk = mk*x + (1-mk)*x_prev  (and xv), fp32 -> bf16
// ---------------------------------------------------------------------------
__global__ void prep_kernel(const float* __restrict__ x,
                            const float* __restrict__ tmrkv,
                            __bf16* __restrict__ xk,
                            __bf16* __restrict__ xv) {
    int i = blockIdx.x * blockDim.x + threadIdx.x;
    const int total = S_DIM * B_DIM * E_DIM;
    if (i >= total) return;
    int e = i & (E_DIM - 1);
    float mk = tmrkv[E_DIM + e];
    float mv = tmrkv[2 * E_DIM + e];
    float xc = x[i];
    const int stride = B_DIM * E_DIM;
    float xp = (i >= stride) ? x[i - stride] : 0.0f;
    xk[i] = (__bf16)(mk * xc + (1.0f - mk) * xp);
    xv[i] = (__bf16)(mv * xc + (1.0f - mv) * xp);
}

// ---------------------------------------------------------------------------
// WMMA GEMM: C(M,N) f32 = A(M,K) bf16 row-major x Bt(N,K) bf16 (B col-major).
// Block tile 128x128, 8 waves arranged 2(M) x 4(N); each wave computes a
// 64x32 macro-tile = 4x2 v_wmma_f32_16x16x32_bf16 tiles per K-step:
// 8 WMMA per 12 b128 loads (A frags reused over 2 N-tiles, B over 4 M-tiles).
//
// Fragment layouts used here (16-bit, wave32; contiguous k order, not the ISA table order):
//   A 16x32: lane L: m = L%16, h = L/16; elems 0..7  = K(k0+16h+0..7)
//                                        elems 8..15 = K(k0+16h+8..15)
//   B 32x16: lane L: n = L%16, h = L/16; elems 0..15 = K(k0+16h+0..15)
//   C 16x16: vgpr r: row = base + 8h + r, col = base + L%16
// ---------------------------------------------------------------------------
__global__ __launch_bounds__(256) void gemm_bf16_wmma_kernel(
    const __bf16* __restrict__ A,
    const __bf16* __restrict__ Bt,
    float* __restrict__ C,
    int M, int N, int K) {
    const int wave = threadIdx.x >> 5;      // 0..7
    const int lane = threadIdx.x & 31;
    const int lmod = lane & 15;
    const int h    = lane >> 4;

    const int block_m = blockIdx.y * 128;
    const int block_n = blockIdx.x * 128;
    const int wm = block_m + (wave >> 2) * 64;   // 2 wave-rows of 64
    const int wn = block_n + (wave & 3) * 32;    // 4 wave-cols of 32

    v8f acc[4][2] = {};

    const __bf16* aptr[4];
    #pragma unroll
    for (int tm = 0; tm < 4; ++tm)
        aptr[tm] = A + (size_t)(wm + 16 * tm + lmod) * K + 16 * h;
    const __bf16* b0 = Bt + (size_t)(wn + lmod) * K + 16 * h;
    const __bf16* b1 = Bt + (size_t)(wn + 16 + lmod) * K + 16 * h;

    for (int k0 = 0; k0 < K; k0 += 32) {
        v16bf bf0 = *(const v16bf*)(b0 + k0);
        v16bf bf1 = *(const v16bf*)(b1 + k0);
        #pragma unroll
        for (int tm = 0; tm < 4; ++tm) {
            v8bf lo = *(const v8bf*)(aptr[tm] + k0);
            v8bf hi = *(const v8bf*)(aptr[tm] + k0 + 8);
            v16bf af = __builtin_shufflevector(lo, hi,
                          0,1,2,3,4,5,6,7,8,9,10,11,12,13,14,15);
            acc[tm][0] = __builtin_amdgcn_wmma_f32_16x16x32_bf16(
                            false, af, false, bf0, (short)0, acc[tm][0],
                            false, false);
            acc[tm][1] = __builtin_amdgcn_wmma_f32_16x16x32_bf16(
                            false, af, false, bf1, (short)0, acc[tm][1],
                            false, false);
        }
    }

    #pragma unroll
    for (int tm = 0; tm < 4; ++tm) {
        int row = wm + 16 * tm + 8 * h;
        #pragma unroll
        for (int r = 0; r < 8; ++r) {
            C[(size_t)(row + r) * N + wn + lmod]      = acc[tm][0][r];
            C[(size_t)(row + r) * N + wn + 16 + lmod] = acc[tm][1][r];
        }
    }
}

// ---------------------------------------------------------------------------
// Parallel WKV scan in 3 passes over 32 chunks of 64 steps.
// Element of the monoid: (a, b, w, p); per-element w == time_decay (constant
// per channel), so a chunk summary's w is implicitly CLEN*w.
// combine(L, R): q = max(p_L + w_R, p_R); e1 = exp(p_L + w_R - q);
//                e2 = exp(p_R - q); a = a_L*e1 + a_R*e2; b = b_L*e1 + b_R*e2
// ---------------------------------------------------------------------------

// Pass 1: per-(channel, chunk) fold -> chunk summary (a, b, p)
__global__ void wkv_chunk_reduce_kernel(const float* __restrict__ kbuf,
                                        const float* __restrict__ vbuf,
                                        const float* __restrict__ time_decay,
                                        float* __restrict__ sumA,
                                        float* __restrict__ sumB,
                                        float* __restrict__ sumP) {
    int idx = blockIdx.x * blockDim.x + threadIdx.x;   // chunk*NCHAN + ch
    if (idx >= NCHAN * NCHUNK) return;
    int ch    = idx & (NCHAN - 1);
    int chunk = idx >> 13;                              // / NCHAN
    int e = ch & (E_DIM - 1);
    const float w = time_decay[e];

    float a = 0.0f, b = 0.0f, p = -1.0e30f;
    int t0 = chunk * CLEN;
    for (int t = t0; t < t0 + CLEN; ++t) {
        size_t off = (size_t)t * NCHAN + ch;
        float kt = kbuf[off];
        float vt = vbuf[off];
        float pw = p + w;
        float q  = fmaxf(pw, kt);
        float e1 = __expf(pw - q);
        float e2 = __expf(kt - q);
        a = a * e1 + vt * e2;
        b = b * e1 + e2;
        p = q;
    }
    sumA[idx] = a;
    sumB[idx] = b;
    sumP[idx] = p;
}

// Pass 2: per-channel exclusive scan over the 32 chunk summaries
__global__ void wkv_chunk_prefix_kernel(const float* __restrict__ sumA,
                                        const float* __restrict__ sumB,
                                        const float* __restrict__ sumP,
                                        const float* __restrict__ time_decay,
                                        float* __restrict__ preA,
                                        float* __restrict__ preB,
                                        float* __restrict__ preP) {
    int ch = blockIdx.x * blockDim.x + threadIdx.x;
    if (ch >= NCHAN) return;
    int e = ch & (E_DIM - 1);
    const float cw = (float)CLEN * time_decay[e];   // chunk's total decay

    float a = 0.0f, b = 0.0f, p = -1.0e30f;         // identity prefix
    for (int j = 0; j < NCHUNK; ++j) {
        int off = j * NCHAN + ch;
        preA[off] = a; preB[off] = b; preP[off] = p;
        float ra = sumA[off], rb = sumB[off], rp = sumP[off];
        float pw = p + cw;
        float q  = fmaxf(pw, rp);
        float e1 = __expf(pw - q);
        float e2 = __expf(rp - q);
        a = a * e1 + ra * e2;
        b = b * e1 + rb * e2;
        p = q;
    }
}

// Pass 3: re-fold each chunk from its prefix, emit rwkv = c/d as bf16
__global__ void wkv_chunk_emit_kernel(const float* __restrict__ kbuf,
                                      const float* __restrict__ vbuf,
                                      const float* __restrict__ time_decay,
                                      const float* __restrict__ time_first,
                                      const float* __restrict__ preA,
                                      const float* __restrict__ preB,
                                      const float* __restrict__ preP,
                                      __bf16* __restrict__ rwkv) {
    int idx = blockIdx.x * blockDim.x + threadIdx.x;   // chunk*NCHAN + ch
    if (idx >= NCHAN * NCHUNK) return;
    int ch    = idx & (NCHAN - 1);
    int chunk = idx >> 13;
    int e = ch & (E_DIM - 1);
    const float w = time_decay[e];
    const float u = time_first[e];

    float a = preA[idx], b = preB[idx], p = preP[idx];
    int t0 = chunk * CLEN;
    for (int t = t0; t < t0 + CLEN; ++t) {
        size_t off = (size_t)t * NCHAN + ch;
        float kt = kbuf[off];
        float vt = vbuf[off];

        // inclusive state: y_t = combine(y_{t-1}, (vt, 1, w, kt))
        float pw = p + w;
        float q  = fmaxf(pw, kt);
        float e1 = __expf(pw - q);
        float e2 = __expf(kt - q);
        a = a * e1 + vt * e2;
        b = b * e1 + e2;
        p = q;

        // output mix: exp_mix_frac(p_state, kt + u + w, a, b, vt, 1)
        float s  = kt + u + w;
        float qq = fmaxf(p, s);
        float x1 = __expf(p - qq);
        float x2 = __expf(s - qq);
        float c  = a * x1 + vt * x2;
        float d  = b * x1 + x2;
        rwkv[off] = (__bf16)(c / d);
    }
}

// ---------------------------------------------------------------------------
// Launch
// ---------------------------------------------------------------------------
extern "C" void kernel_launch(void* const* d_in, const int* in_sizes, int n_in,
                              void* d_out, int out_size, void* d_ws, size_t ws_size,
                              hipStream_t stream) {
    const float* x          = (const float*)d_in[0];
    const float* tmrkv      = (const float*)d_in[1];
    const float* Wk         = (const float*)d_in[2];
    const float* Wv         = (const float*)d_in[3];
    const float* time_decay = (const float*)d_in[4];
    const float* time_first = (const float*)d_in[5];
    const float* Wout       = (const float*)d_in[6];
    float* out = (float*)d_out;

    // Workspace layout (bytes)
    const size_t ME = (size_t)M_DIM * E_DIM;           // 16.7M elems
    const size_t EE = (size_t)E_DIM * E_DIM;           // 1M elems
    const size_t CS = (size_t)NCHAN * NCHUNK * 4;      // 1MB per scan buffer
    char* w = (char*)d_ws;
    __bf16* xk    = (__bf16*)(w);                              // 2*ME
    __bf16* xv    = (__bf16*)(w + 2 * ME);                     // 2*ME
    __bf16* WkT   = (__bf16*)(w + 4 * ME);                     // 2*EE
    __bf16* WvT   = (__bf16*)(w + 4 * ME + 2 * EE);            // 2*EE
    __bf16* WoutT = (__bf16*)(w + 4 * ME + 4 * EE);            // 2*EE
    float*  kbuf  = (float*)(w + 4 * ME + 6 * EE);             // 4*ME
    float*  vbuf  = (float*)(w + 8 * ME + 6 * EE);             // 4*ME
    char*   sbase = w + 12 * ME + 6 * EE;
    float*  sumA  = (float*)(sbase);
    float*  sumB  = (float*)(sbase + CS);
    float*  sumP  = (float*)(sbase + 2 * CS);
    float*  preA  = (float*)(sbase + 3 * CS);
    float*  preB  = (float*)(sbase + 4 * CS);
    float*  preP  = (float*)(sbase + 5 * CS);
    __bf16* rwkv  = xk;    // alias: xk is dead once kbuf is produced

    // 1) weight transpose+convert (3x)
    {
        int total = E_DIM * E_DIM;
        int blocks = (total + 255) / 256;
        wt_convert_kernel<<<blocks, 256, 0, stream>>>(Wk,   WkT,   E_DIM);
        wt_convert_kernel<<<blocks, 256, 0, stream>>>(Wv,   WvT,   E_DIM);
        wt_convert_kernel<<<blocks, 256, 0, stream>>>(Wout, WoutT, E_DIM);
    }

    // 2) token-mix prep
    {
        int total = S_DIM * B_DIM * E_DIM;
        int blocks = (total + 255) / 256;
        prep_kernel<<<blocks, 256, 0, stream>>>(x, tmrkv, xk, xv);
    }

    // 3) k = xk @ Wk ; v = xv @ Wv
    {
        dim3 grid(E_DIM / 128, M_DIM / 128, 1);
        gemm_bf16_wmma_kernel<<<grid, 256, 0, stream>>>(xk, WkT, kbuf,
                                                        M_DIM, E_DIM, E_DIM);
        gemm_bf16_wmma_kernel<<<grid, 256, 0, stream>>>(xv, WvT, vbuf,
                                                        M_DIM, E_DIM, E_DIM);
    }

    // 4) parallel WKV scan: chunk reduce -> chunk prefix -> emit
    {
        int total = NCHAN * NCHUNK;
        int blocks = (total + 255) / 256;
        wkv_chunk_reduce_kernel<<<blocks, 256, 0, stream>>>(
            kbuf, vbuf, time_decay, sumA, sumB, sumP);
        wkv_chunk_prefix_kernel<<<(NCHAN + 255) / 256, 256, 0, stream>>>(
            sumA, sumB, sumP, time_decay, preA, preB, preP);
        wkv_chunk_emit_kernel<<<blocks, 256, 0, stream>>>(
            kbuf, vbuf, time_decay, time_first, preA, preB, preP, rwkv);
    }

    // 5) out = rwkv @ Wout
    {
        dim3 grid(E_DIM / 128, M_DIM / 128, 1);
        gemm_bf16_wmma_kernel<<<grid, 256, 0, stream>>>(rwkv, WoutT, out,
                                                        M_DIM, E_DIM, E_DIM);
    }
}
